// Seq2Seq_46832323395602
// MI455X (gfx1250) — compile-verified, matrix operand order checked by reading, not run on hardware
//
#include <hip/hip_runtime.h>
#include <hip/hip_bf16.h>

typedef __attribute__((ext_vector_type(16))) _Float16 v16h;
typedef __attribute__((ext_vector_type(8)))  float    v8f;

#define Bq 64
#define Tq 400
#define Uq 64
#define Eq 128
#define Hq 400
#define Mq 10

#define NT_G 100      // 1600/16 n-tiles for gate GEMMs
#define KT_G 17       // 544/32 k-tiles (531 valid, padded)
#define NT_O 8        // 128/16 (121 valid)
#define KT_O 25       // 800/32
#define NWG  100
#define BDIM 128

// ---------------- workspace layout (bytes) ----------------
static constexpr size_t SZ_WG   = (size_t)NT_G * KT_G * 512 * 2;   // packed gate weights (f16)
static constexpr size_t SZ_WOC  = (size_t)NT_O * KT_O * 512 * 2;
static constexpr size_t SZ_AF   = (size_t)KT_G * 4 * 32 * 16 * 2;  // A-fragment buffer (f16)
static constexpr size_t SZ_AOF  = (size_t)KT_O * 4 * 32 * 16 * 2;
static constexpr size_t OFF_W0C = 0;
static constexpr size_t OFF_W1C = OFF_W0C + SZ_WG;
static constexpr size_t OFF_W2C = OFF_W1C + SZ_WG;
static constexpr size_t OFF_WOC = OFF_W2C + SZ_WG;
static constexpr size_t OFF_A0  = OFF_WOC + SZ_WOC;
static constexpr size_t OFF_A1  = OFF_A0 + SZ_AF;
static constexpr size_t OFF_A2  = OFF_A1 + SZ_AF;
static constexpr size_t OFF_AO  = OFF_A2 + SZ_AF;
static constexpr size_t OFF_B0  = OFF_AO + SZ_AOF;
static constexpr size_t OFF_B1  = OFF_B0 + 1600 * 4;
static constexpr size_t OFF_B2  = OFF_B1 + 1600 * 4;
static constexpr size_t OFF_BO  = OFF_B2 + 1600 * 4;
static constexpr size_t OFF_HA  = OFF_BO + 128 * 4;
static constexpr size_t OFF_C0S = OFF_HA  + (size_t)Bq * Hq * 4;
static constexpr size_t OFF_C1S = OFF_C0S + (size_t)Bq * Hq * 4;
static constexpr size_t OFF_KST = OFF_C1S + (size_t)Bq * Hq * 4;
static constexpr size_t OFF_CTX = OFF_KST + 1024;
static constexpr size_t OFF_OUT = OFF_CTX + (size_t)Bq * Uq * Eq * 4;
static constexpr size_t OFF_PART= OFF_OUT + (size_t)2 * Bq * 128 * 4;
static constexpr size_t OFF_BAR = OFF_PART + 128 * 4;

// ---------------- device helpers ----------------
__device__ __forceinline__ float sigf(float x) { return 1.f / (1.f + expf(-x)); }

// scatter one f32 value as f16 into the 16x32 A-fragment layout:
// lanes 0-15 hold K 0..7 (e 0..7) and 16..23 (e 8..15); lanes 16-31 hold K 8..15 and 24..31
__device__ __forceinline__ void frag_store(_Float16* base, int m, int k, float v) {
  int kt = k >> 5, kk = k & 31;
  int laneHi = (kk >> 3) & 1;
  int e = ((kk >> 4) << 3) | (kk & 7);
  int lane = (m & 15) | (laneHi << 4);
  base[((((kt << 2) + (m >> 4)) * 32 + lane) << 4) + e] = (_Float16)v;
}

__device__ __forceinline__ v8f gemm_tile(const _Float16* Afrag, const _Float16* Wfrag,
                                         int ktiles, int ntile, int mt, int lane) {
  const v16h* A = (const v16h*)Afrag;
  const v16h* Bv = (const v16h*)Wfrag;
  v8f acc = {};
  int bbase = ntile * ktiles * 32;
  for (int kt = 0; kt < ktiles; ++kt) {
    v16h a = A[(kt * 4 + mt) * 32 + lane];
    v16h b = Bv[bbase + kt * 32 + lane];
    acc = __builtin_amdgcn_wmma_f32_16x16x32_f16(false, a, false, b, (short)0, acc,
                                                 false, false);
  }
  return acc;
}

__device__ __forceinline__ void gridbar(unsigned* cnt, unsigned* gen, int nwg) {
  __syncthreads();
  if (threadIdx.x == 0) {
    __threadfence();
    unsigned g = __hip_atomic_load(gen, __ATOMIC_RELAXED, __HIP_MEMORY_SCOPE_AGENT);
    unsigned a = __hip_atomic_fetch_add(cnt, 1u, __ATOMIC_ACQ_REL, __HIP_MEMORY_SCOPE_AGENT);
    if (a == (unsigned)(nwg - 1)) {
      __hip_atomic_store(cnt, 0u, __ATOMIC_RELAXED, __HIP_MEMORY_SCOPE_AGENT);
      __hip_atomic_fetch_add(gen, 1u, __ATOMIC_RELEASE, __HIP_MEMORY_SCOPE_AGENT);
    } else {
      while (__hip_atomic_load(gen, __ATOMIC_ACQUIRE, __HIP_MEMORY_SCOPE_AGENT) == g)
        __builtin_amdgcn_s_sleep(2);
    }
    __threadfence();
  }
  __syncthreads();
}

// LSTM pointwise epilogue; phase: 1=cell0(hA), 3=cell1, 4=cell2
__device__ __forceinline__ void lstm_pw(v8f acc, int wg, int mt, int lane, const float* bias,
                                        float* smem, int phase, float* c0s, float* c1s,
                                        const float* c2c, float* hA, _Float16* A0f,
                                        _Float16* A1f, _Float16* A2f, _Float16* AOf) {
  float* S = smem + mt * 256;
  int c = lane & 15;
  float bv = bias[wg * 16 + c];
  int rbase = (lane >> 4) << 3;
#pragma unroll
  for (int r = 0; r < 8; ++r) S[(rbase + r) * 16 + c] = acc[r] + bv;
  __syncthreads();
#pragma unroll
  for (int half = 0; half < 2; ++half) {
    int oi = lane + 32 * half;     // 0..63 = 16 rows x 4 units
    int m = oi >> 2, ju = oi & 3;
    float i_ = S[m * 16 + 4 * ju + 0];
    float f_ = S[m * 16 + 4 * ju + 1];
    float g_ = S[m * 16 + 4 * ju + 2];
    float o_ = S[m * 16 + 4 * ju + 3];
    int b = 16 * mt + m;
    int j = 4 * wg + ju;
    float cin = (phase == 1) ? c0s[b * Hq + j] : (phase == 3) ? c1s[b * Hq + j] : c2c[j];
    float cn = sigf(f_) * cin + sigf(i_) * tanhf(g_);
    float h = sigf(o_) * tanhf(cn);
    if (phase == 1) {
      hA[b * Hq + j] = h;                       // feeds attention only
    } else if (phase == 3) {
      c0s[b * Hq + j] = cn;                     // new c0 state = n0c
      frag_store(A2f, b, 131 + j, h);           // cell2 input this step
      frag_store(A0f, b, 131 + j, h);           // h0 state for next step
      frag_store(AOf, b, j, h);                 // out-proj cols 0..399
    } else {
      c1s[b * Hq + j] = cn;                     // new c1 state = n1c
      frag_store(A1f, b, 131 + j, h);           // h1 state for next step
      frag_store(AOf, b, 400 + j, h);           // out-proj cols 400..799
    }
  }
}

__device__ __forceinline__ void loss_row(const float* o, const float* strokes, int b, int tt,
                                         float& accS, float& accE) {
  float ox = strokes[(b * Tq + tt) * 3 + 0];
  float oy = strokes[(b * Tq + tt) * 3 + 1];
  float oe = strokes[(b * Tq + tt) * 3 + 2];
  float pimax = -1e30f;
  for (int k = 0; k < 20; ++k) pimax = fmaxf(pimax, o[80 + k]);
  float pis = 0.f;
  for (int k = 0; k < 20; ++k) pis += expf(o[80 + k] - pimax);
  float logZ = pimax + logf(pis);
  float amax = -1e30f, asum = 0.f;
  for (int k = 0; k < 20; ++k) {
    float mu0 = o[2 * k], mu1 = o[2 * k + 1];
    float ls0 = o[40 + 2 * k], ls1 = o[40 + 2 * k + 1];
    float s0 = expf(ls0), s1 = expf(ls1);
    float rho = tanhf(o[100 + k]);
    float t0 = (ox - mu0) / s0, t1 = (oy - mu1) / s1;
    float Zq = t0 * t0 + t1 * t1 - 2.f * rho * t0 * t1;
    float omr = 1.f - rho * rho;
    float a = (o[80 + k] - logZ) + (-Zq / (2.f * omr)) -
              (1.8378770664093453f + ls0 + ls1 + 0.5f * logf(omr));
    if (a > amax) { asum = asum * expf(amax - a) + 1.f; amax = a; }
    else asum += expf(a - amax);
  }
  accS += -(amax + logf(asum));
  float z = o[120];
  accE += fmaxf(z, 0.f) - z * oe + log1pf(expf(-fabsf(z)));
}

// ---------------- prep kernels ----------------
__global__ void k_zero(unsigned* p, int n) {
  for (int i = blockIdx.x * blockDim.x + threadIdx.x; i < n; i += gridDim.x * blockDim.x)
    p[i] = 0u;
}

// pack weights into B-fragment layout [ntile][ktile][lane][e] (gate-interleaved rows)
__global__ void k_pack(char* ws, size_t off, const float* S1, int s1s, int K1,
                       const float* S2, int s2s, int Kvalid, int ktiles, int ntiles,
                       int nvalid, int gi) {
  _Float16* dst = (_Float16*)(ws + off);
  int idx = blockIdx.x * blockDim.x + threadIdx.x;
  int total = ntiles * ktiles * 512;
  if (idx >= total) return;
  int e = idx & 15, lane = (idx >> 4) & 31;
  int kt = (idx >> 9) % ktiles, nt = (idx >> 9) / ktiles;
  int n = nt * 16 + (lane & 15);
  int kk = ((e >> 3) << 4) | ((lane >> 4) << 3) | (e & 7);
  int k = kt * 32 + kk;
  float v = 0.f;
  if (n < nvalid && k < Kvalid) {
    int row = gi ? ((n & 3) * Hq + (n >> 2)) : n;
    v = (k < K1) ? S1[row * s1s + k] : S2[row * s2s + (k - K1)];
  }
  dst[idx] = (_Float16)v;
}

// effective biases (gate-interleaved) with constant-h2 folds
__global__ void k_bias(char* ws, const float* bih0, const float* bhh0, const float* bih1,
                       const float* bhh1, const float* Wih1, const float* bih2,
                       const float* bhh2, const float* Whh2, const float* Wo,
                       const float* bo, const float* h0in) {
  float* b0 = (float*)(ws + OFF_B0);
  float* b1 = (float*)(ws + OFF_B1);
  float* b2 = (float*)(ws + OFF_B2);
  float* bO = (float*)(ws + OFF_BO);
  int n = blockIdx.x * blockDim.x + threadIdx.x;
  if (n < 1600) {
    int row = (n & 3) * Hq + (n >> 2);
    b0[n] = bih0[row] + bhh0[row];
    float s1 = 0.f, s2 = 0.f;
    for (int j = 0; j < Hq; ++j) {
      float h2 = h0in[2 * Hq + j];
      s1 += Wih1[row * 531 + 131 + j] * h2;
      s2 += Whh2[row * Hq + j] * h2;
    }
    b1[n] = bih1[row] + bhh1[row] + s1;
    b2[n] = bih2[row] + bhh2[row] + s2;
  } else if (n < 1728) {
    int m = n - 1600;
    float v = 0.f;
    if (m < 121) {
      v = bo[m];
      for (int j = 0; j < Hq; ++j) v += Wo[m * 1200 + 800 + j] * h0in[2 * Hq + j];
    }
    bO[m] = v;
  }
}

__global__ void k_ctx(char* ws, const float* emb, const int* chars, const float* mask) {
  float* ctx = (float*)(ws + OFF_CTX);
  int idx = blockIdx.x * blockDim.x + threadIdx.x;
  if (idx >= Bq * Uq * Eq) return;
  int b = idx / (Uq * Eq);
  int u = (idx / Eq) % Uq;
  int e = idx % Eq;
  ctx[idx] = emb[chars[b * Uq + u] * Eq + e] * mask[b * Uq + u];
}

// initial states and A0 staging for t=0
__global__ void k_init(char* ws, const float* strokes, const float* c0in, const float* k0in,
                       const float* w0in, const float* h0in) {
  float* c0s = (float*)(ws + OFF_C0S);
  float* c1s = (float*)(ws + OFF_C1S);
  float* kst = (float*)(ws + OFF_KST);
  _Float16* A0f = (_Float16*)(ws + OFF_A0);
  int idx = blockIdx.x * blockDim.x + threadIdx.x;
  int NS = Bq * Hq;
  if (idx < NS) { c0s[idx] = c0in[idx % Hq]; return; }
  idx -= NS;
  if (idx < NS) { c1s[idx] = c0in[Hq + idx % Hq]; return; }
  idx -= NS;
  if (idx < Bq * Mq) { kst[idx] = k0in[idx % Mq]; return; }
  idx -= Bq * Mq;
  if (idx < Bq * 531) {
    int b = idx / 531, k = idx % 531;
    float v = (k < 3) ? strokes[(b * Tq + 0) * 3 + k]
                      : (k < 131) ? w0in[k - 3] : h0in[k - 131];
    frag_store(A0f, b, k, v);
  }
}

// ---------------- persistent recurrent kernel ----------------
__global__ __launch_bounds__(BDIM) void k_rnn(char* ws, const float* strokes, const float* Wa,
                                              const float* ba, const float* c0in, float* dout) {
  const int wg = blockIdx.x, tid = threadIdx.x;
  const int lane = tid & 31, mt = tid >> 5;
  _Float16* W0c = (_Float16*)(ws + OFF_W0C);
  _Float16* W1c = (_Float16*)(ws + OFF_W1C);
  _Float16* W2c = (_Float16*)(ws + OFF_W2C);
  _Float16* WOc = (_Float16*)(ws + OFF_WOC);
  _Float16* A0f = (_Float16*)(ws + OFF_A0);
  _Float16* A1f = (_Float16*)(ws + OFF_A1);
  _Float16* A2f = (_Float16*)(ws + OFF_A2);
  _Float16* AOf = (_Float16*)(ws + OFF_AO);
  float* b0 = (float*)(ws + OFF_B0);
  float* b1 = (float*)(ws + OFF_B1);
  float* b2 = (float*)(ws + OFF_B2);
  float* bO = (float*)(ws + OFF_BO);
  float* hA = (float*)(ws + OFF_HA);
  float* c0s = (float*)(ws + OFF_C0S);
  float* c1s = (float*)(ws + OFF_C1S);
  float* kst = (float*)(ws + OFF_KST);
  float* ctx = (float*)(ws + OFF_CTX);
  float* outbuf = (float*)(ws + OFF_OUT);
  float* part = (float*)(ws + OFF_PART);
  unsigned* barc = (unsigned*)(ws + OFF_BAR);
  unsigned* barg = barc + 1;
  const float* c2c = c0in + 2 * Hq;
  __shared__ float smem[1024];
  float accS = 0.f, accE = 0.f;

  for (int t = 0; t < Tq; ++t) {
    // ---- Phase 1: cell0 GEMM + pointwise (hA); overlap loss(t-1) ----
    {
      v8f d = gemm_tile(A0f, W0c, KT_G, wg, mt, lane);
      lstm_pw(d, wg, mt, lane, b0, smem, 1, c0s, c1s, c2c, hA, A0f, A1f, A2f, AOf);
    }
    if (wg < 32 && tid < 2 && t > 0) {
      int b = wg * 2 + tid;
      loss_row(outbuf + ((t - 1) & 1) * Bq * 128 + b * 128, strokes, b, t - 1, accS, accE);
    }
    gridbar(barc, barg, NWG);

    // ---- Phase 2: attention window (WGs 0..7, 8 batch rows each); WG8 stages x_t ----
    if (wg < 8) {
      float* sp = smem;           // [8][32] exp(hA@Wa^T+ba)
      float* skap = smem + 256;   // [8][16] kappa
      float* sphi = smem + 384;   // [8][64] phi
      for (int idx = tid; idx < 8 * 30; idx += BDIM) {
        int r = idx / 30, o = idx % 30;
        const float* hrow = hA + (wg * 8 + r) * Hq;
        const float* wrow = Wa + o * Hq;
        float s = ba[o];
        for (int k = 0; k < Hq; ++k) s += hrow[k] * wrow[k];
        sp[r * 32 + o] = expf(s);
      }
      __syncthreads();
      for (int idx = tid; idx < 8 * Mq; idx += BDIM) {
        int r = idx / Mq, m2 = idx % Mq;
        int b = wg * 8 + r;
        float kap = sp[r * 32 + 20 + m2] + kst[b * Mq + m2];
        kst[b * Mq + m2] = kap;
        skap[r * 16 + m2] = kap;
      }
      __syncthreads();
      for (int idx = tid; idx < 8 * Uq; idx += BDIM) {
        int r = idx / Uq, u = idx % Uq;
        float ph = 0.f;
        for (int m2 = 0; m2 < Mq; ++m2) {
          float dd = skap[r * 16 + m2] - (float)u;
          ph += sp[r * 32 + m2] * expf(-sp[r * 32 + 10 + m2] * dd * dd);
        }
        sphi[r * Uq + u] = ph;
      }
      __syncthreads();
      for (int idx = tid; idx < 8 * Eq; idx += BDIM) {
        int r = idx / Eq, e = idx % Eq;
        int b = wg * 8 + r;
        const float* cb = ctx + (size_t)b * Uq * Eq + e;
        float w = 0.f;
        for (int u = 0; u < Uq; ++u) w += sphi[r * Uq + u] * cb[u * Eq];
        frag_store(A0f, b, 3 + e, w);   // w state for next step's cell0
        frag_store(A1f, b, 3 + e, w);   // this step's cell1/cell2 inputs
        frag_store(A2f, b, 3 + e, w);
      }
    } else if (wg == 8) {
      for (int idx = tid; idx < Bq * 3; idx += BDIM) {
        int b = idx / 3, c = idx % 3;
        float v = strokes[(b * Tq + t) * 3 + c];
        frag_store(A1f, b, c, v);
        frag_store(A2f, b, c, v);
      }
    }
    gridbar(barc, barg, NWG);

    // ---- Phase 3: cell1 GEMM + pointwise (n0h/n0c) ----
    {
      v8f d = gemm_tile(A1f, W1c, KT_G, wg, mt, lane);
      lstm_pw(d, wg, mt, lane, b1, smem, 3, c0s, c1s, c2c, hA, A0f, A1f, A2f, AOf);
    }
    gridbar(barc, barg, NWG);

    // ---- Phase 4: cell2 GEMM + pointwise (n1h/n1c) ----
    {
      v8f d = gemm_tile(A2f, W2c, KT_G, wg, mt, lane);
      lstm_pw(d, wg, mt, lane, b2, smem, 4, c0s, c1s, c2c, hA, A0f, A1f, A2f, AOf);
    }
    gridbar(barc, barg, NWG);

    // ---- Phase 5: output projection; WG8 pre-stages x_{t+1} into A0 ----
    if (wg < NT_O) {
      v8f d = gemm_tile(AOf, WOc, KT_O, wg, mt, lane);
      int n = wg * 16 + (lane & 15);
      if (n < 121) {
        float bv = bO[n];
        float* ob = outbuf + (t & 1) * Bq * 128;
        int rbase = (lane >> 4) << 3;
#pragma unroll
        for (int r = 0; r < 8; ++r) {
          int b = 16 * mt + rbase + r;
          ob[b * 128 + n] = d[r] + bv;
        }
      }
    } else if (wg == 8 && t + 1 < Tq) {
      for (int idx = tid; idx < Bq * 3; idx += BDIM) {
        int b = idx / 3, c = idx % 3;
        frag_store(A0f, b, c, strokes[(b * Tq + (t + 1)) * 3 + c]);
      }
    }
    gridbar(barc, barg, NWG);
  }

  // ---- tail: loss(T-1), ordered deterministic reduction ----
  if (wg < 32 && tid < 2) {
    int b = wg * 2 + tid;
    loss_row(outbuf + ((Tq - 1) & 1) * Bq * 128 + b * 128, strokes, b, Tq - 1, accS, accE);
    part[b * 2 + 0] = accS;
    part[b * 2 + 1] = accE;
  }
  gridbar(barc, barg, NWG);
  if (wg == 0 && tid == 0) {
    float S = 0.f, E2 = 0.f;
    for (int b = 0; b < Bq; ++b) { S += part[2 * b]; E2 += part[2 * b + 1]; }
    dout[0] = S / (float)(Bq * Tq);
    dout[1] = E2 / (float)(Bq * Tq);
  }
}

// ---------------- host ----------------
extern "C" void kernel_launch(void* const* d_in, const int* in_sizes, int n_in,
                              void* d_out, int out_size, void* d_ws, size_t ws_size,
                              hipStream_t stream) {
  (void)in_sizes; (void)n_in; (void)out_size; (void)ws_size;
  char* ws = (char*)d_ws;
  const float* strokes = (const float*)d_in[0];
  const int*   chars   = (const int*)d_in[1];
  const float* mask    = (const float*)d_in[2];
  const float* emb     = (const float*)d_in[3];
  const float* W_ih0   = (const float*)d_in[4];
  const float* W_hh0   = (const float*)d_in[5];
  const float* b_ih0   = (const float*)d_in[6];
  const float* b_hh0   = (const float*)d_in[7];
  const float* W_ih1   = (const float*)d_in[8];
  const float* W_hh1   = (const float*)d_in[9];
  const float* b_ih1   = (const float*)d_in[10];
  const float* b_hh1   = (const float*)d_in[11];
  const float* W_ih2   = (const float*)d_in[12];
  const float* W_hh2   = (const float*)d_in[13];
  const float* b_ih2   = (const float*)d_in[14];
  const float* b_hh2   = (const float*)d_in[15];
  const float* Wa      = (const float*)d_in[16];
  const float* ba      = (const float*)d_in[17];
  const float* Wo      = (const float*)d_in[18];
  const float* bo      = (const float*)d_in[19];
  const float* h0      = (const float*)d_in[20];
  const float* c0      = (const float*)d_in[21];
  const float* w0      = (const float*)d_in[22];
  const float* k0      = (const float*)d_in[23];

  // zero activation-fragment buffers + partials + barrier
  int zw1 = (int)((SZ_AF * 3 + SZ_AOF) / 4);
  k_zero<<<(zw1 + 255) / 256, 256, 0, stream>>>((unsigned*)(ws + OFF_A0), zw1);
  k_zero<<<1, 256, 0, stream>>>((unsigned*)(ws + OFF_PART), (int)((128 * 4 + 64) / 4));

  // states + A0 staging for t=0
  int initN = Bq * Hq * 2 + Bq * Mq + Bq * 531;
  k_init<<<(initN + 255) / 256, 256, 0, stream>>>(ws, strokes, c0, k0, w0, h0);

  // pack weights (f16, WMMA B-fragment layout, gate-interleaved)
  int gT = NT_G * KT_G * 512, oT = NT_O * KT_O * 512;
  k_pack<<<(gT + 255) / 256, 256, 0, stream>>>(ws, OFF_W0C, W_ih0, 131, 131, W_hh0, Hq,
                                               531, KT_G, NT_G, 1600, 1);
  k_pack<<<(gT + 255) / 256, 256, 0, stream>>>(ws, OFF_W1C, W_ih1, 531, 131, W_hh1, Hq,
                                               531, KT_G, NT_G, 1600, 1);
  k_pack<<<(gT + 255) / 256, 256, 0, stream>>>(ws, OFF_W2C, W_ih2, 531, 531, W_ih2, 531,
                                               531, KT_G, NT_G, 1600, 1);
  k_pack<<<(oT + 255) / 256, 256, 0, stream>>>(ws, OFF_WOC, Wo, 1200, 800, Wo, 1200,
                                               800, KT_O, NT_O, 121, 0);

  // effective biases (fold constant h2 paths) and attention context
  k_bias<<<(1728 + 255) / 256, 256, 0, stream>>>(ws, b_ih0, b_hh0, b_ih1, b_hh1, W_ih1,
                                                 b_ih2, b_hh2, W_hh2, Wo, bo, h0);
  k_ctx<<<(Bq * Uq * Eq + 255) / 256, 256, 0, stream>>>(ws, emb, chars, mask);

  // persistent recurrent kernel
  k_rnn<<<NWG, BDIM, 0, stream>>>(ws, strokes, Wa, ba, c0, (float*)d_out);
}
